// Gemma2FusedAttention_58652073394234
// MI455X (gfx1250) — matrix operand order repaired and checked by reading, not compiled, not run on hardware
//
#include <hip/hip_runtime.h>
#include <hip/hip_bf16.h>

// ---------------------------------------------------------------------------
// Gemma2 fused attention for MI455X (gfx1250, wave32, WMMA).
// All matmuls run on V_WMMA_F32_16X16X32_BF16 (f32 accumulate).
// ---------------------------------------------------------------------------

typedef __bf16 bf16_t;
typedef __attribute__((ext_vector_type(16))) __bf16 v16bf;
typedef __attribute__((ext_vector_type(8)))  __bf16 v8bf;
typedef __attribute__((ext_vector_type(8)))  float  v8f;

namespace cfg {
constexpr int S      = 4096;
constexpr int HID    = 2304;
constexpr int NH     = 8;
constexpr int NKV    = 4;
constexpr int HD     = 256;
constexpr int QKV_N  = (NH + 2 * NKV) * HD;  // 4096
constexpr int O_K    = NH * HD;              // 2048
constexpr int SW     = 2048;                 // sliding window
constexpr float SOFTCAP  = 50.0f;
constexpr float SCALE    = 0.0625f;          // 256^-0.5
constexpr float MASK_MIN = -1.0e9f;
}

// ---------------------------------------------------------------------------
// Fragment helpers (ISA 7.12.2 wave32 layouts).
// A (16x32 bf16): lane = {half = l>>4, m = l&15}; elements e<8 -> K = half*8+e,
//                 e>=8 -> K = 16 + half*8 + (e-8).  Two contiguous 16B runs.
// B (32x16 bf16): lane = {half = l>>4, n = l&15}; same K order as A: e<8 -> K = half*8+e,
//                 e>=8 -> K = 16 + half*8 + (e-8).  Two contiguous 16B runs (B stored K-contiguous, i.e.
//                 row-major [N][K] for D = A * B with B[k][n] = Mat[n][k]).
// C/D (16x16 f32): vgpr r, lane l -> row = r + 8*(l>>4), col = l&15.
// ---------------------------------------------------------------------------

__device__ __forceinline__ v16bf cat8(v8bf lo, v8bf hi) {
  return __builtin_shufflevector(lo, hi, 0, 1, 2, 3, 4, 5, 6, 7,
                                         8, 9, 10, 11, 12, 13, 14, 15);
}

__device__ __forceinline__ v16bf load_a16x32(const bf16_t* __restrict__ base,
                                             int row0, int ld, int k0, int lane) {
  const int m = lane & 15, h = lane >> 4;
  const bf16_t* p = base + (size_t)(row0 + m) * ld + k0 + h * 8;
  v8bf lo = *(const v8bf*)p;
  v8bf hi = *(const v8bf*)(p + 16);
  return cat8(lo, hi);
}

__device__ __forceinline__ v16bf load_b32x16(const bf16_t* __restrict__ base,
                                             int col0, int ld, int k0, int lane) {
  const int n = lane & 15, h = lane >> 4;
  const bf16_t* p = base + (size_t)(col0 + n) * ld + k0 + h * 8;
  v8bf lo = *(const v8bf*)p;
  v8bf hi = *(const v8bf*)(p + 16);
  return cat8(lo, hi);
}

__device__ __forceinline__ v8f wmma_bf16(v16bf a, v16bf b, v8f c) {
  return __builtin_amdgcn_wmma_f32_16x16x32_bf16(false, a, false, b,
                                                 (short)0, c, false, false);
}

// ---------------------------------------------------------------------------
// fp32 -> bf16 staging
// ---------------------------------------------------------------------------
__global__ __launch_bounds__(256) void cvt_f32_bf16(const float* __restrict__ in,
                                                    bf16_t* __restrict__ out,
                                                    long n) {
  long i = (long)blockIdx.x * 256 + threadIdx.x;
  const long stride = (long)gridDim.x * 256;
  for (; i < n; i += stride) out[i] = (bf16_t)in[i];
}

// ---------------------------------------------------------------------------
// C[M,N] = A[M,K] * B[N,K]^T   (A,B bf16; C fp32 or bf16)
// Block: 256 threads = 8 waves as 2(M) x 4(N); block tile 64 x 256;
// each wave computes a 32x64 tile (8 WMMA accumulators).
// Requires M%64==0, N%256==0, K%32==0 (true for all uses here).
// ---------------------------------------------------------------------------
__global__ __launch_bounds__(256) void gemm_bf16(const bf16_t* __restrict__ A,
                                                 const bf16_t* __restrict__ B,
                                                 float* __restrict__ Cf,
                                                 bf16_t* __restrict__ Cb,
                                                 int M, int N, int K,
                                                 int store_bf) {
  const int lane = threadIdx.x & 31;
  const int w    = threadIdx.x >> 5;
  const int row0 = blockIdx.x * 64 + (w >> 2) * 32;
  const int col0 = blockIdx.y * 256 + (w & 3) * 64;
  const int nn = lane & 15, hf = lane >> 4;

  v8f zero = {};
  v8f acc[2][4];
#pragma unroll
  for (int i = 0; i < 2; ++i)
#pragma unroll
    for (int t = 0; t < 4; ++t) acc[i][t] = zero;

  for (int k0 = 0; k0 < K; k0 += 32) {
    if (k0 + 32 < K) {
      __builtin_prefetch((const void*)(B + (size_t)(col0 + nn) * K + k0 + 32), 0, 0);
      __builtin_prefetch((const void*)(A + (size_t)(row0 + nn) * K + k0 + 32), 0, 0);
    }
    v16bf a0 = load_a16x32(A, row0, K, k0, lane);
    v16bf a1 = load_a16x32(A, row0 + 16, K, k0, lane);
#pragma unroll
    for (int t = 0; t < 4; ++t) {
      v16bf b = load_b32x16(B, col0 + t * 16, K, k0, lane);
      acc[0][t] = wmma_bf16(a0, b, acc[0][t]);
      acc[1][t] = wmma_bf16(a1, b, acc[1][t]);
    }
  }

#pragma unroll
  for (int i = 0; i < 2; ++i)
#pragma unroll
    for (int t = 0; t < 4; ++t) {
      const int n  = col0 + t * 16 + nn;
      const int mb = row0 + i * 16 + 8 * hf;
#pragma unroll
      for (int r = 0; r < 8; ++r) {
        const float v = acc[i][t][r];
        if (store_bf) Cb[(size_t)(mb + r) * N + n] = (bf16_t)v;
        else          Cf[(size_t)(mb + r) * N + n] = v;
      }
    }
}

// ---------------------------------------------------------------------------
// RoPE + head split.  qkv row layout: [Q 0..2047 | K 2048..3071 | V 3072..4095].
// Writes Qr[h][s][d], Kr[kv][s][d] (rotary applied), Vt[kv][d][s] (transposed
// so the PV-stage B fragment is a contiguous per-lane load).
// grid = (S, 16): y<8 Q heads, y<12 K heads, else V heads.  block = 256 (= HD).
// ---------------------------------------------------------------------------
__global__ __launch_bounds__(256) void rope_split(const bf16_t* __restrict__ qkv,
                                                  bf16_t* __restrict__ Qr,
                                                  bf16_t* __restrict__ Kr,
                                                  bf16_t* __restrict__ Vt) {
  using namespace cfg;
  const int t = blockIdx.x;
  const int y = blockIdx.y;
  const int d = threadIdx.x;  // 0..255
  const bf16_t* row = qkv + (size_t)t * QKV_N;

  if (y < 12) {
    const int base = (y < 8) ? y * HD : NH * HD + (y - 8) * HD;
    const float x = (float)row[base + d];
    const int dm = d & 127;
    // inv_freq = 10000^(-2*dm/256) ; ln(10000) = 9.210340371976184
    const float inv = __expf(-9.210340371976184f * ((float)(2 * dm) * (1.0f / 256.0f)));
    const float ang = (float)t * inv;
    const float c = cosf(ang), sn = sinf(ang);
    const float partner = (float)row[base + ((d < 128) ? d + 128 : d - 128)];
    const float outv = (d < 128) ? (x * c - partner * sn) : (x * c + partner * sn);
    if (y < 8) Qr[((size_t)y * S + t) * HD + d]        = (bf16_t)outv;
    else       Kr[((size_t)(y - 8) * S + t) * HD + d]  = (bf16_t)outv;
  } else {
    const int kv = y - 12;
    Vt[((size_t)kv * HD + d) * S + t] = row[(NH + NKV) * HD + kv * HD + d];
  }
}

// ---------------------------------------------------------------------------
// Flash attention with sliding-window mask (computed analytically), softcap,
// online softmax.  Block = 128 threads (4 waves); block owns 64 query rows of
// one head; each wave owns 16 query rows x full 256 head-dim (16 f32 WMMA
// accumulators).  Key tiles of 64 within the window (<=33 tiles).
// ---------------------------------------------------------------------------
__global__ __launch_bounds__(128) void attn_fused(const bf16_t* __restrict__ Qr,
                                                  const bf16_t* __restrict__ Kr,
                                                  const bf16_t* __restrict__ Vt,
                                                  bf16_t* __restrict__ AttnO) {
  using namespace cfg;
  __shared__ __attribute__((aligned(32))) bf16_t plds[4][16 * 64];

  const int h  = blockIdx.y;
  const int kv = h >> 1;  // GQA: groups = 2
  const int q0 = blockIdx.x * 64;
  const int w    = threadIdx.x >> 5;
  const int lane = threadIdx.x & 31;
  const int qrow0 = q0 + w * 16;
  const int nn = lane & 15, half = lane >> 4;

  const bf16_t* Q  = Qr + (size_t)h * S * HD;
  const bf16_t* Kh = Kr + (size_t)kv * S * HD;
  const bf16_t* Vh = Vt + (size_t)kv * HD * S;
  bf16_t* pl = &plds[w][0];

  v8f zero = {};
  v8f o[16];
#pragma unroll
  for (int th = 0; th < 16; ++th) o[th] = zero;
  float mrow[8], lrow[8];
#pragma unroll
  for (int r = 0; r < 8; ++r) { mrow[r] = -3.0e38f; lrow[r] = 0.0f; }

  const int jmin = (q0 - (SW - 1)) > 0 ? (q0 - (SW - 1)) : 0;
  const int ktlo = jmin >> 6;
  const int kthi = (q0 + 63) >> 6;

  for (int kt = ktlo; kt <= kthi; ++kt) {
    const int k0keys = kt * 64;

    // ---- S strip = Q(16x256) * K^T(256x64) ----
    v8f sacc[4];
#pragma unroll
    for (int t = 0; t < 4; ++t) sacc[t] = zero;
    for (int hk = 0; hk < 8; ++hk) {
      v16bf qa = load_a16x32(Q, qrow0, HD, hk * 32, lane);
#pragma unroll
      for (int t = 0; t < 4; ++t) {
        v16bf kb = load_b32x16(Kh, k0keys + t * 16, HD, hk * 32, lane);
        sacc[t] = wmma_bf16(qa, kb, sacc[t]);
      }
    }

    // ---- scale, softcap, analytic sliding-window causal mask ----
    float rowmax[8];
#pragma unroll
    for (int r = 0; r < 8; ++r) rowmax[r] = -3.0e38f;
#pragma unroll
    for (int t = 0; t < 4; ++t)
#pragma unroll
      for (int r = 0; r < 8; ++r) {
        float s = sacc[t][r] * SCALE;
        s = tanhf(s * (1.0f / SOFTCAP)) * SOFTCAP;
        const int i = qrow0 + r + 8 * half;
        const int j = k0keys + t * 16 + nn;
        if (!((j <= i) && (j > i - SW))) s += MASK_MIN;
        sacc[t][r] = s;
        rowmax[r] = fmaxf(rowmax[r], s);
      }
    // row max across the 16 lanes that share each row group
#pragma unroll
    for (int off = 1; off < 16; off <<= 1)
#pragma unroll
      for (int r = 0; r < 8; ++r)
        rowmax[r] = fmaxf(rowmax[r], __shfl_xor(rowmax[r], off, 32));

    float alpha[8];
#pragma unroll
    for (int r = 0; r < 8; ++r) {
      const float mn = fmaxf(mrow[r], rowmax[r]);
      alpha[r] = __expf(mrow[r] - mn);
      mrow[r] = mn;
    }

    // ---- P = exp(s - m); stash bf16 P in LDS (C-layout -> A-layout bridge) ----
    float rs[8];
#pragma unroll
    for (int r = 0; r < 8; ++r) rs[r] = 0.0f;
#pragma unroll
    for (int t = 0; t < 4; ++t)
#pragma unroll
      for (int r = 0; r < 8; ++r) {
        const float p = __expf(sacc[t][r] - mrow[r]);
        rs[r] += p;
        pl[(r + 8 * half) * 64 + t * 16 + nn] = (bf16_t)p;
      }
#pragma unroll
    for (int off = 1; off < 16; off <<= 1)
#pragma unroll
      for (int r = 0; r < 8; ++r)
        rs[r] += __shfl_xor(rs[r], off, 32);
#pragma unroll
    for (int r = 0; r < 8; ++r) lrow[r] = lrow[r] * alpha[r] + rs[r];

    // rescale running O
#pragma unroll
    for (int th = 0; th < 16; ++th)
#pragma unroll
      for (int r = 0; r < 8; ++r) o[th][r] *= alpha[r];

    // ---- O += P(16x64) * V(64x256)  (same-wave LDS ops are in-order) ----
#pragma unroll
    for (int kk = 0; kk < 2; ++kk) {
      const bf16_t* ap = pl + nn * 64 + kk * 32 + half * 8;
      v8bf lo = *(const v8bf*)ap;
      v8bf hi = *(const v8bf*)(ap + 16);
      v16bf pa = cat8(lo, hi);
#pragma unroll
      for (int th = 0; th < 16; ++th) {
        v16bf vb = load_b32x16(Vh, th * 16, S, k0keys + kk * 32, lane);
        o[th] = wmma_bf16(pa, vb, o[th]);
      }
    }
  }

  // ---- normalize and emit bf16 [token][h*256+d] for the output projection ----
#pragma unroll
  for (int r = 0; r < 8; ++r) lrow[r] = 1.0f / lrow[r];
#pragma unroll
  for (int th = 0; th < 16; ++th) {
    const int col = h * HD + th * 16 + nn;
#pragma unroll
    for (int r = 0; r < 8; ++r) {
      const int token = qrow0 + r + 8 * half;
      AttnO[(size_t)token * O_K + col] = (bf16_t)(o[th][r] * lrow[r]);
    }
  }
}

// ---------------------------------------------------------------------------
// Host-side launcher.
// inputs: 0 hidden_states f32[1,4096,2304], 1 attention_mask (unused, computed
// analytically), 2 position_ids (unused, = arange), 3 Wqkv f32[4096,2304],
// 4 Wo f32[2304,2048].  output: f32[1,4096,2304].
// ---------------------------------------------------------------------------
extern "C" void kernel_launch(void* const* d_in, const int* in_sizes, int n_in,
                              void* d_out, int out_size, void* d_ws, size_t ws_size,
                              hipStream_t stream) {
  using namespace cfg;
  const float* hs   = (const float*)d_in[0];
  const float* Wqkv = (const float*)d_in[3];
  const float* Wo   = (const float*)d_in[4];
  float* out = (float*)d_out;

  char* w = (char*)d_ws;
  size_t off = 0;
  bf16_t* Xbf   = (bf16_t*)(w + off);            // S*HID bf16 (18.9 MB)
  bf16_t* AttnO = Xbf;                           // alias: Xbf dead before attn writes
  off += (size_t)S * HID * sizeof(bf16_t);
  bf16_t* Wqkvb = (bf16_t*)(w + off); off += (size_t)QKV_N * HID * sizeof(bf16_t);
  bf16_t* Wob   = (bf16_t*)(w + off); off += (size_t)HID * O_K * sizeof(bf16_t);
  bf16_t* QKVb  = (bf16_t*)(w + off); off += (size_t)S * QKV_N * sizeof(bf16_t);
  bf16_t* Qr    = (bf16_t*)(w + off); off += (size_t)NH * S * HD * sizeof(bf16_t);
  bf16_t* Kr    = (bf16_t*)(w + off); off += (size_t)NKV * S * HD * sizeof(bf16_t);
  bf16_t* Vt    = (bf16_t*)(w + off); off += (size_t)NKV * HD * S * sizeof(bf16_t);
  (void)ws_size; (void)in_sizes; (void)n_in; (void)out_size;

  // 1) stage operands to bf16
  cvt_f32_bf16<<<2048, 256, 0, stream>>>(hs,   Xbf,   (long)S * HID);
  cvt_f32_bf16<<<2048, 256, 0, stream>>>(Wqkv, Wqkvb, (long)QKV_N * HID);
  cvt_f32_bf16<<<2048, 256, 0, stream>>>(Wo,   Wob,   (long)HID * O_K);

  // 2) QKV projection: [4096,2304] x [4096,2304]^T -> bf16 [4096,4096]
  gemm_bf16<<<dim3(S / 64, QKV_N / 256), 256, 0, stream>>>(
      Xbf, Wqkvb, nullptr, QKVb, S, QKV_N, HID, 1);

  // 3) RoPE + head split (+ V transpose)
  rope_split<<<dim3(S, 16), 256, 0, stream>>>(QKVb, Qr, Kr, Vt);

  // 4) sliding-window flash attention
  attn_fused<<<dim3(S / 64, NH), 128, 0, stream>>>(Qr, Kr, Vt, AttnO);

  // 5) output projection: [4096,2048] x [2304,2048]^T -> f32 out
  gemm_bf16<<<dim3(S / 64, HID / 256), 256, 0, stream>>>(
      AttnO, Wob, out, nullptr, S, HID, O_K, 0);
}
